// SNNforauditory_33784212750539
// MI455X (gfx1250) — matrix operand order repaired and checked by reading, not compiled, not run on hardware
//
#include <hip/hip_runtime.h>

typedef __attribute__((ext_vector_type(16))) _Float16 v16h;
typedef __attribute__((ext_vector_type(8)))  float    v8f;

// ---------------------------------------------------------------------------
// B-fragment (16-bit, KxN = 32x16) for V_WMMA_F32_16X16X32_F16, wave32 layout:
//   lanes 0-15  : v0-3 -> K=0..7, v4-7 -> K=16..23 (even K [15:0], odd K [31:16]), col n = lane
//   lanes 16-31 : v0-3 -> K=8..15, v4-7 -> K=24..31, col n = lane-16
// B = W^T, i.e. element (k, n) = W[n][k]; pads with 0 outside (in_f, out_f).
// Built once per wave; weights are tiny and L2-resident.
// ---------------------------------------------------------------------------
__device__ inline v16h build_bfrag(const float* __restrict__ W, int out_f,
                                   int in_f, int ncol_base, int lane) {
  const int n = (lane & 15) + ncol_base;
  const int kbase = (lane >> 4) << 3;
  v16h r;
#pragma unroll
  for (int v = 0; v < 8; ++v) {
    const int k0 = ((v & 4) ? 16 : 0) + kbase + 2 * (v & 3);
    const int k1 = k0 + 1;
    float f0 = 0.0f, f1 = 0.0f;
    if (n < out_f) {
      if (k0 < in_f) f0 = W[n * in_f + k0];
      if (k1 < in_f) f1 = W[n * in_f + k1];
    }
    r[2 * v]     = (_Float16)f0;
    r[2 * v + 1] = (_Float16)f1;
  }
  return r;
}

// ---------------------------------------------------------------------------
// A-fragment (16-bit, MxK = 16x32) from a row-major LDS tile [16][NCOLS] f16:
//   lanes 0-15  (m = lane):    v0-3 -> K=0..7,   v4-7 -> K=16..23
//   lanes 16-31 (m = lane-16): v0-3 -> K=8..15,  v4-7 -> K=24..31
// NCOLS==16: v<4 is always in range (k<=14), v>=4 always out (k>=16) -> the
// zero-padding decision is compile-time; no runtime predication.
// Pairs are K-adjacent, so loads merge into ds_load_b128.
// ---------------------------------------------------------------------------
template <int NCOLS>
__device__ inline v16h load_afrag(const _Float16* __restrict__ lds, int lane) {
  const int m = lane & 15;
  const int g = (lane >> 4) * 8;
  v16h a;
#pragma unroll
  for (int v = 0; v < 8; ++v) {
    if (NCOLS == 32 || v < 4) {
      const int k = ((v & 4) ? 16 : 0) + g + 2 * (v & 3);
      a[2 * v]     = lds[m * NCOLS + k];
      a[2 * v + 1] = lds[m * NCOLS + k + 1];
    } else {
      a[2 * v]     = (_Float16)0.0f;
      a[2 * v + 1] = (_Float16)0.0f;
    }
  }
  return a;
}

// Threshold f32 accumulator at 2.0 (== v_threshold * tau) and store spikes
// into the wave-private LDS tile in row-major order. D-layout: lane holds
// col n = lane&15, VGPR v holds row m = (lane>=16 ? 8 : 0) + v.
template <int NCOLS>
__device__ inline void store_spikes(_Float16* __restrict__ lds, int ncol_off,
                                    int lane, v8f acc) {
  const int n  = (lane & 15) + ncol_off;
  const int mb = (lane >> 4) * 8;
#pragma unroll
  for (int v = 0; v < 8; ++v)
    lds[(mb + v) * NCOLS + n] = (acc[v] >= 2.0f) ? (_Float16)1.0f
                                                 : (_Float16)0.0f;
}

__global__ __launch_bounds__(256) void snn_auditory_wmma_kernel(
    const float* __restrict__ x,  const float* __restrict__ W1,
    const float* __restrict__ W2, const float* __restrict__ W3,
    const float* __restrict__ W4, float* __restrict__ out, int nrows) {
  __shared__ _Float16 spk[8][16 * 32];  // 1 KB spike staging per wave
  __shared__ float    ost[8][160];      // 640 B output staging per wave
  const int lane = threadIdx.x & 31;
  const int wave = threadIdx.x >> 5;
  _Float16* lds  = spk[wave];
  float*    osts = ost[wave];

  // Weight fragments, built once, held in VGPRs across the whole loop.
  const v16h B1a = build_bfrag(W1, 32,  5,  0, lane);
  const v16h B1b = build_bfrag(W1, 32,  5, 16, lane);
  const v16h B2a = build_bfrag(W2, 32, 32,  0, lane);
  const v16h B2b = build_bfrag(W2, 32, 32, 16, lane);
  const v16h B3  = build_bfrag(W3, 16, 32,  0, lane);
  const v16h B4  = build_bfrag(W4, 10, 16,  0, lane);

  const int waves_per_blk = blockDim.x >> 5;
  const int wave_id = blockIdx.x * waves_per_blk + wave;
  const int nwaves  = gridDim.x * waves_per_blk;
  const int ntiles  = nrows >> 4;  // 16 rows per tile

  const int  m  = lane & 15;
  const bool lo = lane < 16;  // lanes 0-15 own K=0..7 of the A fragment
  const v8f  zero = {};

  for (int t = wave_id; t < ntiles; t += nwaves) {
    const int row0 = t << 4;
    if (t + nwaves < ntiles)  // global_prefetch_b8 for the next tile's input
      __builtin_prefetch(&x[(size_t)(t + nwaves) * 80], 0, 0);

    // ---- Layer 1: A = x tile [16 x 5] (K zero-padded to 32), f32 -> f16.
    // All 5 addresses are valid for every lane (lanes 16-31 share m with
    // lanes 0-15), so load unconditionally -> no EXEC divergence; zero the
    // upper half-wave's elements with plain selects.
    const float* xrow = x + (size_t)(row0 + m) * 5;
    const float e0 = xrow[0], e1 = xrow[1], e2 = xrow[2], e3 = xrow[3],
                e4 = xrow[4];
    v16h a;
#pragma unroll
    for (int i = 0; i < 16; ++i) a[i] = (_Float16)0.0f;
    a[0] = lo ? (_Float16)e0 : (_Float16)0.0f;
    a[1] = lo ? (_Float16)e1 : (_Float16)0.0f;
    a[2] = lo ? (_Float16)e2 : (_Float16)0.0f;
    a[3] = lo ? (_Float16)e3 : (_Float16)0.0f;
    a[4] = lo ? (_Float16)e4 : (_Float16)0.0f;

    v8f acc_a = __builtin_amdgcn_wmma_f32_16x16x32_f16(false, a, false, B1a,
                                                       (short)0, zero, false, false);
    v8f acc_b = __builtin_amdgcn_wmma_f32_16x16x32_f16(false, a, false, B1b,
                                                       (short)0, zero, false, false);
    store_spikes<32>(lds,  0, lane, acc_a);
    store_spikes<32>(lds, 16, lane, acc_b);

    // ---- Layer 2: [16x32] @ W2^T[32x32] ----
    a = load_afrag<32>(lds, lane);
    acc_a = __builtin_amdgcn_wmma_f32_16x16x32_f16(false, a, false, B2a,
                                                   (short)0, zero, false, false);
    acc_b = __builtin_amdgcn_wmma_f32_16x16x32_f16(false, a, false, B2b,
                                                   (short)0, zero, false, false);
    store_spikes<32>(lds,  0, lane, acc_a);
    store_spikes<32>(lds, 16, lane, acc_b);

    // ---- Layer 3: [16x32] @ W3^T[32x16] ----
    a = load_afrag<32>(lds, lane);
    acc_a = __builtin_amdgcn_wmma_f32_16x16x32_f16(false, a, false, B3,
                                                   (short)0, zero, false, false);
    store_spikes<16>(lds, 0, lane, acc_a);

    // ---- Layer 4: [16x16 padded to K=32] @ W4^T[16x10 padded] ----
    a = load_afrag<16>(lds, lane);
    acc_a = __builtin_amdgcn_wmma_f32_16x16x32_f16(false, a, false, B4,
                                                   (short)0, zero, false, false);

    // Spike + stage f32 output in LDS (D-layout -> row-major [16][10]) ...
    const int n = lane & 15;
    if (n < 10) {
      const int mb = (lane >> 4) * 8;
#pragma unroll
      for (int v = 0; v < 8; ++v)
        osts[(mb + v) * 10 + n] = (acc_a[v] >= 2.0f) ? 1.0f : 0.0f;
    }
    // ... then write the tile's contiguous 640 B fully coalesced:
    // 160 dwords == 5 per lane, every lane active, no predication.
    float* otile = out + (size_t)row0 * 10;
#pragma unroll
    for (int i = 0; i < 5; ++i)
      otile[lane + 32 * i] = osts[lane + 32 * i];
  }
}

extern "C" void kernel_launch(void* const* d_in, const int* in_sizes, int n_in,
                              void* d_out, int out_size, void* d_ws, size_t ws_size,
                              hipStream_t stream) {
  (void)n_in; (void)out_size; (void)d_ws; (void)ws_size;
  const float* x  = (const float*)d_in[0];
  const float* W1 = (const float*)d_in[1];
  const float* W2 = (const float*)d_in[2];
  const float* W3 = (const float*)d_in[3];
  const float* W4 = (const float*)d_in[4];
  float* out = (float*)d_out;

  const int nrows = in_sizes[0] / 5;  // B = 2,097,152
  // 1024 blocks x 8 waves = 8192 waves; 131072 tiles -> 16 tiles/wave.
  dim3 grid(1024), block(256);
  snn_auditory_wmma_kernel<<<grid, block, 0, stream>>>(x, W1, W2, W3, W4, out,
                                                       nrows);
}
